// SBLS_38594576122396
// MI455X (gfx1250) — compile-verified, matrix operand order checked by reading, not run on hardware
//
#include <hip/hip_runtime.h>

// ---------------------------------------------------------------------------
// SBLS fused SNN pipeline for gfx1250 (wave32, WMMA 16x16x32 f16 -> f32 acc)
// ---------------------------------------------------------------------------

typedef __attribute__((ext_vector_type(16))) _Float16 v16h;
typedef __attribute__((ext_vector_type(8)))  _Float16 v8h;
typedef __attribute__((ext_vector_type(8)))  float    v8f;

#define IN_DIM    784
#define KX        800        // IN_DIM padded to multiple of 32
#define NFEAT     100
#define NFEAT_PAD 112        // feature cols padded to multiple of 16
#define KF        128        // NFEAT padded to multiple of 32 (main GEMM K)
#define MP_TOT    2000       // MW*MP
#define MP_PAD    2016       // padded to multiple of 16
#define S_STEPS   30
#define NROWS     2048
#define OUTC      2100       // NFEAT + MP_TOT
#define BETA      0.9f
#define THRESH    1.0f

// f16-element offsets inside workspace (total ~19.7 MB; all 16B aligned)
#define XH_OFF   0
#define W1H_OFF  (NROWS * KX)                         // Xh: 2048 x 800
#define W2H_OFF  (W1H_OFF + NFEAT_PAD * KX)           // W1h^T: 112 x 800
#define ZH_OFF   (W2H_OFF + MP_PAD * KF)              // W2h^T: 2016 x 128
                                                      // Zh: 30 x 2048 x 128

// ---------------------------------------------------------------------------
// WMMA fragment loader: 16-bit A/B fragment, 16x32 tile from row-major [row][k]
// Per ISA 7.12.2: lanes 0-15 -> rows 0-15 with K 0..7 & 16..23,
//                 lanes 16-31 -> rows 0-15 with K 8..15 & 24..31.
// base points at the tile's first row; ldk is the row stride in f16 elements.
// ---------------------------------------------------------------------------
__device__ __forceinline__ v16h load_frag(const _Float16* base, int ldk, int k0) {
    int lane = threadIdx.x & 31;
    int row  = lane & 15;
    int koff = k0 + ((lane >> 4) << 3);          // 0 or 8
    const _Float16* p = base + row * ldk + koff;
    v8h lo = *(const v8h*)(p);                    // K = koff .. koff+7
    v8h hi = *(const v8h*)(p + 16);               // K = koff+16 .. koff+23
    return __builtin_shufflevector(lo, hi, 0,1,2,3,4,5,6,7,8,9,10,11,12,13,14,15);
}

__device__ __forceinline__ v8f wmma_f16(v16h a, v16h b, v8f c) {
    return __builtin_amdgcn_wmma_f32_16x16x32_f16(
        /*neg_a=*/false, a, /*neg_b=*/false, b,
        /*c_mod=*/(short)0, c, /*reuse_a=*/false, /*reuse_b=*/false);
}

// ---------------------------------------------------------------------------
// Prep kernels: f32 -> f16 convert, pad, transpose weights; zero spike pad
// ---------------------------------------------------------------------------
__global__ void zero_zh_kernel(_Float16* zh) {
    const size_t n = (size_t)S_STEPS * NROWS * KF / 8;    // v8h stores
    v8h z = {};
    for (size_t i = (size_t)blockIdx.x * blockDim.x + threadIdx.x; i < n;
         i += (size_t)gridDim.x * blockDim.x)
        ((v8h*)zh)[i] = z;
}

__global__ void prep_x_kernel(const float* X, _Float16* Xh) {
    const int n = NROWS * KX;
    for (int i = blockIdx.x * blockDim.x + threadIdx.x; i < n;
         i += gridDim.x * blockDim.x) {
        int r = i / KX, k = i - r * KX;
        Xh[i] = (k < IN_DIM) ? (_Float16)X[r * IN_DIM + k] : (_Float16)0.0f;
    }
}

__global__ void prep_w1_kernel(const float* W1, _Float16* W1h) {
    const int n = NFEAT_PAD * KX;                 // store transposed: [p][k]
    for (int i = blockIdx.x * blockDim.x + threadIdx.x; i < n;
         i += gridDim.x * blockDim.x) {
        int p = i / KX, k = i - p * KX;
        float v = (p < NFEAT && k < IN_DIM) ? W1[k * NFEAT + p] : 0.0f;
        W1h[i] = (_Float16)v;
    }
}

__global__ void prep_w2_kernel(const float* W2, _Float16* W2h) {
    const int n = MP_PAD * KF;                    // store transposed: [p][k]
    for (int i = blockIdx.x * blockDim.x + threadIdx.x; i < n;
         i += gridDim.x * blockDim.x) {
        int p = i / KF, k = i - p * KF;
        float v = (p < MP_TOT && k < NFEAT) ? W2[k * MP_TOT + p] : 0.0f;
        W2h[i] = (_Float16)v;
    }
}

// ---------------------------------------------------------------------------
// Feature kernel: Z = sigmoid((X@W1 + B1)/3); spikes = (U < Z); writes f16
// spike tensor Zh[s][n][k] and Z-rate mean into out[:, 0:100].
// One wave per 16x16 tile; 128 row-tiles x 7 col-tiles = 896 waves.
// ---------------------------------------------------------------------------
__global__ __launch_bounds__(256) void feat_kernel(
        const float* __restrict__ B1, const float* __restrict__ U,
        const _Float16* __restrict__ Xh, const _Float16* __restrict__ W1h,
        _Float16* __restrict__ Zh, float* __restrict__ out) {
    const int wave   = threadIdx.x >> 5;
    const int tileId = blockIdx.x * 8 + wave;     // 0..895
    const int ntile  = tileId / 7;
    const int ptile  = tileId - ntile * 7;
    const int m0 = ntile * 16;
    const int n0 = ptile * 16;

    const _Float16* Abase = Xh  + (size_t)m0 * KX;
    const _Float16* Bbase = W1h + (size_t)n0 * KX;

    v8f c = {};
    #pragma unroll 5
    for (int k0 = 0; k0 < KX; k0 += 32) {
        v16h a = load_frag(Abase, KX, k0);
        v16h b = load_frag(Bbase, KX, k0);
        c = wmma_f16(a, b, c);
    }

    const int lane  = threadIdx.x & 31;
    const int n     = n0 + (lane & 15);           // feature index
    const int mbase = m0 + ((lane >> 4) << 3);
    const bool valid = (n < NFEAT);
    const float b1 = valid ? B1[n] : 0.0f;

    float zprob[8];
    float cnt[8];
    #pragma unroll
    for (int i = 0; i < 8; ++i) {
        float h = c[i] + b1;
        zprob[i] = 1.0f / (1.0f + __expf(-h * (1.0f / 3.0f)));
        cnt[i] = 0.0f;
    }

    for (int s = 0; s < S_STEPS; ++s) {
        #pragma unroll
        for (int i = 0; i < 8; ++i) {
            int m = mbase + i;
            if (valid) {
                float u = U[((size_t)s * NROWS + m) * NFEAT + n];
                float spk = (u < zprob[i]) ? 1.0f : 0.0f;
                cnt[i] += spk;
                Zh[((size_t)s * NROWS + m) * KF + n] = (_Float16)spk;
            }
        }
    }

    if (valid) {
        #pragma unroll
        for (int i = 0; i < 8; ++i)
            out[(size_t)(mbase + i) * OUTC + n] = cnt[i] * (1.0f / S_STEPS);
    }
}

// ---------------------------------------------------------------------------
// Fused enhancement GEMM + LIF + rate kernel.
// One wave per 16x32 output tile (2 N-subtiles): the 4 A fragments streamed
// per timestep now feed 8 WMMAs instead of 4, halving A-side L2 traffic.
// mem/spk/cnt live in VGPRs across all 30 timesteps (no H_pre in memory).
// 128 row-tiles x 63 col-pair-tiles = 8064 waves -> 1008 blocks of 8 waves.
// ---------------------------------------------------------------------------
__global__ __launch_bounds__(256) void lif_kernel(
        const _Float16* __restrict__ Zh, const _Float16* __restrict__ W2h,
        const float* __restrict__ B2, float* __restrict__ out) {
    const int wave   = threadIdx.x >> 5;
    const int tileId = blockIdx.x * 8 + wave;     // 0..8063
    const int ntile  = tileId / 63;
    const int pt2    = tileId - ntile * 63;
    const int m0 = ntile * 16;
    const int p0 = pt2 * 32;

    // B fragments (W2^T tiles) are invariant over timesteps: load once.
    v16h bf[2][4];
    #pragma unroll
    for (int t = 0; t < 2; ++t) {
        const _Float16* Bbase = W2h + (size_t)(p0 + 16 * t) * KF;
        #pragma unroll
        for (int j = 0; j < 4; ++j)
            bf[t][j] = load_frag(Bbase, KF, 32 * j);
    }

    const int lane  = threadIdx.x & 31;
    const int mbase = m0 + ((lane >> 4) << 3);
    int   pcol[2];
    float b2[2];
    #pragma unroll
    for (int t = 0; t < 2; ++t) {
        pcol[t] = p0 + 16 * t + (lane & 15);
        b2[t]   = (pcol[t] < MP_TOT) ? B2[pcol[t]] : 0.0f;
    }

    float mem[2][8], spk[2][8], cnt[2][8];
    #pragma unroll
    for (int t = 0; t < 2; ++t)
        #pragma unroll
        for (int i = 0; i < 8; ++i) { mem[t][i] = 0.0f; spk[t][i] = 0.0f; cnt[t][i] = 0.0f; }

    for (int s = 0; s < S_STEPS; ++s) {
        const _Float16* Abase = Zh + ((size_t)s * NROWS + m0) * KF;
        if (s + 1 < S_STEPS) {   // warm next timestep's spike rows (L2 -> L0)
            const _Float16* nx = Zh + ((size_t)(s + 1) * NROWS + m0 + (lane & 15)) * KF;
            __builtin_prefetch(nx, 0, 3);
        }
        v16h a0 = load_frag(Abase, KF, 0);
        v16h a1 = load_frag(Abase, KF, 32);
        v16h a2 = load_frag(Abase, KF, 64);
        v16h a3 = load_frag(Abase, KF, 96);

        v8f c0 = {};
        c0 = wmma_f16(a0, bf[0][0], c0);
        c0 = wmma_f16(a1, bf[0][1], c0);
        c0 = wmma_f16(a2, bf[0][2], c0);
        c0 = wmma_f16(a3, bf[0][3], c0);
        v8f c1 = {};
        c1 = wmma_f16(a0, bf[1][0], c1);
        c1 = wmma_f16(a1, bf[1][1], c1);
        c1 = wmma_f16(a2, bf[1][2], c1);
        c1 = wmma_f16(a3, bf[1][3], c1);

        // LIF (reset-by-subtraction), previous-step spike feeds the reset
        #pragma unroll
        for (int i = 0; i < 8; ++i) {
            float h0 = c0[i] + b2[0];
            mem[0][i] = BETA * mem[0][i] + h0 - spk[0][i] * THRESH;
            spk[0][i] = (mem[0][i] > THRESH) ? 1.0f : 0.0f;
            cnt[0][i] += spk[0][i];
            float h1 = c1[i] + b2[1];
            mem[1][i] = BETA * mem[1][i] + h1 - spk[1][i] * THRESH;
            spk[1][i] = (mem[1][i] > THRESH) ? 1.0f : 0.0f;
            cnt[1][i] += spk[1][i];
        }
    }

    #pragma unroll
    for (int t = 0; t < 2; ++t) {
        if (pcol[t] < MP_TOT) {
            #pragma unroll
            for (int i = 0; i < 8; ++i)
                out[(size_t)(mbase + i) * OUTC + NFEAT + pcol[t]] = cnt[t][i] * (1.0f / S_STEPS);
        }
    }
}

// ---------------------------------------------------------------------------
// Launch
// ---------------------------------------------------------------------------
extern "C" void kernel_launch(void* const* d_in, const int* in_sizes, int n_in,
                              void* d_out, int out_size, void* d_ws, size_t ws_size,
                              hipStream_t stream) {
    (void)in_sizes; (void)n_in; (void)out_size; (void)ws_size;
    const float* X  = (const float*)d_in[0];   // (2048, 784)
    const float* W1 = (const float*)d_in[1];   // (784, 100)
    const float* B1 = (const float*)d_in[2];   // (100,)
    const float* W2 = (const float*)d_in[3];   // (100, 2000)
    const float* B2 = (const float*)d_in[4];   // (2000,)
    const float* U  = (const float*)d_in[5];   // (30, 2048, 100)
    float* out = (float*)d_out;                // (2048, 2100)

    _Float16* ws  = (_Float16*)d_ws;
    _Float16* Xh  = ws + XH_OFF;
    _Float16* W1h = ws + W1H_OFF;
    _Float16* W2h = ws + W2H_OFF;
    _Float16* Zh  = ws + ZH_OFF;

    // prep: convert/pad/transpose, zero spike-pad region (K 100..127 stays 0)
    zero_zh_kernel<<<1024, 256, 0, stream>>>(Zh);
    prep_x_kernel <<<6400, 256, 0, stream>>>(X,  Xh);
    prep_w1_kernel<<<350,  256, 0, stream>>>(W1, W1h);
    prep_w2_kernel<<<1008, 256, 0, stream>>>(W2, W2h);

    // feature layer + Bernoulli rate coding + Z-rate output (cols 0..99)
    feat_kernel<<<112, 256, 0, stream>>>(B1, U, Xh, W1h, Zh, out);

    // fused 30-step GEMM + LIF + rate output (cols 100..2099)
    lif_kernel<<<1008, 256, 0, stream>>>(Zh, W2h, B2, out);
}
